// StaticQuantLinear_85349590106243
// MI455X (gfx1250) — 8-bit matrix operand order repaired and checked by reading, compiled, not run on hardware
//
#include <hip/hip_runtime.h>

typedef int v8i __attribute__((ext_vector_type(8)));
typedef int v4i __attribute__((ext_vector_type(4)));

#define IN_F   8192
#define OUT_F  32768
#define KSTEPS 128
#define CHUNK  32
#define NCHUNK 4
#define PF     4

__device__ __forceinline__ int unpack_pair(int d0, int d1) {
  unsigned w = __builtin_amdgcn_perm((unsigned)d1, (unsigned)d0, 0x04040000u);
  unsigned u = (w & 0x000F000Fu) | ((w & 0xF000F000u) >> 4);
  u ^= 0x88888888u;
  u -= 0x08080808u;
  u ^= 0x80808080u;
  return (int)u;
}

__global__ __launch_bounds__(256) void quant_pack_a(
    const float* __restrict__ x, const float* __restrict__ act_scale,
    unsigned* __restrict__ abuf) {
  int u = blockIdx.x * 256 + threadIdx.x;
  int v = u & 7;
  int l = (u >> 3) & 31;
  int s = (u >> 8) & 127;
  int t = u >> 15;
  int m = t * 16 + (l & 15);
  int kb = s * 64 + ((l & 16) ? 16 : 0) + ((v >> 2) << 5) + ((v & 3) << 2);
  float inv = 1.0f / act_scale[0];
  const float* xr = x + (size_t)m * IN_F + kb;
  unsigned out = 0;
#pragma unroll
  for (int i = 0; i < 4; ++i) {
    float q = rintf(xr[i] * inv);
    q = fminf(fmaxf(q, -127.0f), 127.0f);
    out |= ((unsigned)((int)q & 0xFF)) << (8 * i);
  }
  abuf[u] = out;
}

__global__ __launch_bounds__(256) void gemm_w4a8(
    const int* __restrict__ wp, const unsigned* __restrict__ abuf,
    const float* __restrict__ wscale, const float* __restrict__ act_scale,
    const float* __restrict__ bias, float* __restrict__ out) {
  __shared__ v4i a_lds[4096];

  const int tid  = threadIdx.x;
  const int lane = tid & 31;
  const int wv   = tid >> 5;
  const int hi   = (lane >> 4) & 1;
  const int o    = blockIdx.x * 128 + wv * 16 + (lane & 15);

  const v4i* wrow = ((const v4i*)wp) + (size_t)o * 1024 + hi * 2;
  const v4i* ab   = (const v4i*)abuf;

  v4i wbuf[PF][4];
#pragma unroll
  for (int j = 0; j < PF; ++j) {
    const v4i* bb = wrow + j * 8;
    wbuf[j][0] = __builtin_nontemporal_load(bb + 0);
    wbuf[j][1] = __builtin_nontemporal_load(bb + 1);
    wbuf[j][2] = __builtin_nontemporal_load(bb + 4);
    wbuf[j][3] = __builtin_nontemporal_load(bb + 5);
  }

  v8i acc0 = {0, 0, 0, 0, 0, 0, 0, 0};
  v8i acc1 = {0, 0, 0, 0, 0, 0, 0, 0};

  for (int c = 0; c < NCHUNK; ++c) {
    __syncthreads();
#pragma unroll
    for (int i = 0; i < 16; ++i) {
      int p = tid + i * 256;
      a_lds[p] = ab[(p >> 11) * 8192 + c * 2048 + (p & 2047)];
    }
    __syncthreads();

#pragma unroll 4
    for (int s2 = 0; s2 < CHUNK; ++s2) {
      const int s = c * CHUNK + s2;
      const int j = s & (PF - 1);
      v4i w0 = wbuf[j][0], w1 = wbuf[j][1], w2 = wbuf[j][2], w3 = wbuf[j][3];
      int sp = s + PF; if (sp > KSTEPS - 1) sp = KSTEPS - 1;
      const v4i* bb = wrow + sp * 8;
      wbuf[j][0] = __builtin_nontemporal_load(bb + 0);
      wbuf[j][1] = __builtin_nontemporal_load(bb + 1);
      wbuf[j][2] = __builtin_nontemporal_load(bb + 4);
      wbuf[j][3] = __builtin_nontemporal_load(bb + 5);

      v8i B;
      B[0] = unpack_pair(w0.x, w0.y);  B[1] = unpack_pair(w0.z, w0.w);
      B[2] = unpack_pair(w1.x, w1.y);  B[3] = unpack_pair(w1.z, w1.w);
      B[4] = unpack_pair(w2.x, w2.y);  B[5] = unpack_pair(w2.z, w2.w);
      B[6] = unpack_pair(w3.x, w3.y);  B[7] = unpack_pair(w3.z, w3.w);

      int ia0 = (s2 * 32 + lane) * 2;
      int ia1 = ((CHUNK + s2) * 32 + lane) * 2;
      v4i a0l = a_lds[ia0], a0h = a_lds[ia0 + 1];
      v4i a1l = a_lds[ia1], a1h = a_lds[ia1 + 1];
      v8i A0 = __builtin_shufflevector(a0l, a0h, 0, 1, 2, 3, 4, 5, 6, 7);
      v8i A1 = __builtin_shufflevector(a1l, a1h, 0, 1, 2, 3, 4, 5, 6, 7);

      acc0 = __builtin_amdgcn_wmma_i32_16x16x64_iu8(true, A0, true, B, acc0,
                                                    false, false);
      acc1 = __builtin_amdgcn_wmma_i32_16x16x64_iu8(true, A1, true, B, acc1,
                                                    false, false);
    }
  }

  const float sc = act_scale[0] * wscale[o];
  const float bo = bias[o];
  const int mrow = hi ? 8 : 0;
#pragma unroll
  for (int r = 0; r < 8; ++r) {
    int m0 = r + mrow;
    out[(size_t)m0 * OUT_F + o]        = (float)acc0[r] * sc + bo;
    out[(size_t)(16 + m0) * OUT_F + o] = (float)acc1[r] * sc + bo;
  }
}

extern "C" void kernel_launch(void* const* d_in, const int* in_sizes, int n_in,
                              void* d_out, int out_size, void* d_ws, size_t ws_size,
                              hipStream_t stream) {
  const float* x      = (const float*)d_in[0];
  const int*   wpack  = (const int*)d_in[1];
  const float* wscale = (const float*)d_in[2];
  const float* ascale = (const float*)d_in[3];
  const float* bias   = (const float*)d_in[4];
  float* y = (float*)d_out;
  unsigned* abuf = (unsigned*)d_ws;

  quant_pack_a<<<256, 256, 0, stream>>>(x, ascale, abuf);
  gemm_w4a8<<<256, 256, 0, stream>>>(wpack, abuf, wscale, ascale, bias, y);
}
